// SearchTransfer_63728724738621
// MI455X (gfx1250) — matrix operand order repaired and checked by reading, not compiled, not run on hardware
//
#include <hip/hip_runtime.h>

typedef __attribute__((ext_vector_type(16))) _Float16 v16h;
typedef __attribute__((ext_vector_type(8)))  _Float16 v8h;
typedef __attribute__((ext_vector_type(8)))  float    v8f;

#define LPATCH 4096      // Ho*Wo = 64*64 patches per batch at every level
#define NBATCH 2
#define LSUPER 128       // l rows per block (supertile)
#define NCHUNK 32        // LPATCH / LSUPER
#define KP 256           // K panel (halves) staged in LDS
#define KPP (KP + 8)     // +16B row pad -> 528B stride, conflict-free frags

// ---------------------------------------------------------------------------
// unfold(img)[b, d, m] with d = c*k*k + r*k + q, m = i*64 + j
// ---------------------------------------------------------------------------
__device__ __forceinline__ float unfold_val(const float* __restrict__ img,
                                            int b, int C, int k, int pad,
                                            int stride, int Himg, int d, int m) {
  int kk = k * k;
  int c = d / kk;
  int r = (d / k) % k;
  int q = d % k;
  int i = m >> 6, j = m & 63;
  int h = i * stride + r - pad;
  int w = j * stride + q - pad;
  if ((unsigned)h >= (unsigned)Himg || (unsigned)w >= (unsigned)Himg) return 0.0f;
  return img[(((size_t)b * C + c) * Himg + h) * Himg + w];
}

// ---------------------------------------------------------------------------
// Pass 1a: per (b,d) row stats of unfolded refsr (normalize axis = L).
// ---------------------------------------------------------------------------
__global__ __launch_bounds__(256) void rs_stats_kernel(
    const float* __restrict__ img, int C, int k, int pad, int stride, int Himg,
    float* __restrict__ rownorm, float* __restrict__ scal) {
  int D = C * k * k;
  int bd = blockIdx.x;
  int b = bd / D, d = bd % D;
  float s = 0.f, ss = 0.f;
  for (int m = threadIdx.x; m < LPATCH; m += 256) {
    float v = unfold_val(img, b, C, k, pad, stride, Himg, d, m);
    s += v; ss += v * v;
  }
  __shared__ float rs[256], rq[256];
  rs[threadIdx.x] = s; rq[threadIdx.x] = ss;
  __syncthreads();
  for (int off = 128; off > 0; off >>= 1) {
    if (threadIdx.x < off) {
      rs[threadIdx.x] += rs[threadIdx.x + off];
      rq[threadIdx.x] += rq[threadIdx.x + off];
    }
    __syncthreads();
  }
  if (threadIdx.x == 0) {
    float nm = fmaxf(sqrtf(rq[0]), 1e-12f);
    rownorm[bd] = nm;
    atomicAdd(&scal[0], rs[0] / nm);
    atomicAdd(&scal[1], rq[0] / (nm * nm));
  }
}

// ---------------------------------------------------------------------------
// Pass 1b: per (b,m) column stats of unfolded lrsr (normalize axis = D).
// ---------------------------------------------------------------------------
__global__ __launch_bounds__(256) void lr_stats_kernel(
    const float* __restrict__ img, int C, int k, int pad, int stride, int Himg,
    float* __restrict__ colnorm, float* __restrict__ scal) {
  int D = C * k * k;
  int bm = blockIdx.x;
  int b = bm >> 12, m = bm & 4095;
  float s = 0.f, ss = 0.f;
  for (int d = threadIdx.x; d < D; d += 256) {
    float v = unfold_val(img, b, C, k, pad, stride, Himg, d, m);
    s += v; ss += v * v;
  }
  __shared__ float rs[256], rq[256];
  rs[threadIdx.x] = s; rq[threadIdx.x] = ss;
  __syncthreads();
  for (int off = 128; off > 0; off >>= 1) {
    if (threadIdx.x < off) {
      rs[threadIdx.x] += rs[threadIdx.x + off];
      rq[threadIdx.x] += rq[threadIdx.x + off];
    }
    __syncthreads();
  }
  if (threadIdx.x == 0) {
    float nm = fmaxf(sqrtf(rq[0]), 1e-12f);
    colnorm[bm] = nm;
    atomicAdd(&scal[2], rs[0] / nm);
    atomicAdd(&scal[3], rq[0] / (nm * nm));
  }
}

// ---------------------------------------------------------------------------
// Pass 2: write mean-subtracted operands as f16, patch-major (L x D).
// which_norm: 0 -> row norm (refsr side, xm), 1 -> col norm (lrsr side, ym)
// ---------------------------------------------------------------------------
__global__ __launch_bounds__(256) void write_panel_kernel(
    const float* __restrict__ img, int C, int k, int pad, int stride, int Himg,
    const float* __restrict__ norms, const float* __restrict__ scal, float Ninv,
    int which_norm, _Float16* __restrict__ panel) {
  int D = C * k * k;
  int bl = blockIdx.x;                // b*4096 + l
  int b = bl >> 12, l = bl & 4095;
  float mean = scal[0] * Ninv;        // mean of rs for this level
  float cn = (which_norm == 1) ? norms[bl] : 0.0f;
  _Float16* row = panel + (size_t)bl * D;
  for (int d = threadIdx.x; d < D; d += 256) {
    float v = unfold_val(img, b, C, k, pad, stride, Himg, d, l);
    float nm = (which_norm == 0) ? norms[b * D + d] : cn;
    v = v / nm - mean;
    row[d] = (_Float16)v;
  }
}

// ---------------------------------------------------------------------------
// Async DMA: global -> LDS, 16B per lane, tracked by ASYNCcnt.
// ---------------------------------------------------------------------------
__device__ __forceinline__ void async_b128(unsigned lds_off, const void* gaddr) {
  asm volatile("global_load_async_to_lds_b128 %0, %1, off"
               :: "v"(lds_off), "v"(gaddr) : "memory");
}
__device__ __forceinline__ void wait_async0() {
  asm volatile("s_wait_asynccnt 0x0" ::: "memory");
}

// Stage `rows` x KP halves from gbase (leading dim D halves) into a padded LDS
// panel (row stride KPP halves). 32 x 16B chunks per row, 256 threads.
__device__ __forceinline__ void stage_panel(_Float16* pan,
                                            const _Float16* __restrict__ gbase,
                                            int D, int kp, int tid, int rows) {
  int total = rows * 32;
  for (int id = tid; id < total; id += 256) {
    int row = id >> 5;
    int cid = id & 31;
    const _Float16* g = gbase + (size_t)row * D + (size_t)kp * KP + cid * 8;
    unsigned lds = (unsigned)(size_t)(pan + row * KPP + cid * 8);
    async_b128(lds, (const void*)g);
  }
}

// ---------------------------------------------------------------------------
// LDS fragment loads matching the CDNA5 16-bit A/B lane layouts.
// A (16x32): lane m<16 holds K{0..7,16..23}; lane m+16 holds K{8..15,24..31}.
// B (32x16): lane n<16 holds col n K{0..7,16..23}; lane n+16 holds col n K{8..15,24..31}.
// ---------------------------------------------------------------------------
__device__ __forceinline__ v16h lds_a_frag(const _Float16* pan, int rowbase,
                                           int kk, int lane) {
  int row = rowbase + (lane & 15);
  int hi = lane >> 4;
  const _Float16* p = pan + row * KPP + kk + hi * 8;
  v8h lo = *(const v8h*)(p);
  v8h hh = *(const v8h*)(p + 16);
  v16h r;
#pragma unroll
  for (int t = 0; t < 8; ++t) { r[t] = lo[t]; r[t + 8] = hh[t]; }
  return r;
}

__device__ __forceinline__ v16h lds_b_frag(const _Float16* pan, int colbase,
                                           int kk, int lane) {
  int col = colbase + (lane & 15);
  int hi = lane >> 4;
  const _Float16* p = pan + col * KPP + kk + hi * 8;
  v8h lo = *(const v8h*)(p);
  v8h hh = *(const v8h*)(p + 16);
  v16h r;
#pragma unroll
  for (int t = 0; t < 8; ++t) { r[t] = lo[t]; r[t + 8] = hh[t]; }
  return r;
}

__device__ __forceinline__ v8f wmma_f16(v16h a, v16h b, v8f c) {
  return __builtin_amdgcn_wmma_f32_16x16x32_f16(false, a, false, b, (short)0, c,
                                                false, false);
}

// ---------------------------------------------------------------------------
// Fused GEMM + row-max:  R[l,m] = sum_d XMT[l,d]*YMT[m,d]
// Block = 256 thr (8 waves), supertile 128(l) x 64(m); each wave owns a
// 32x32 block (4 WMMA accumulators). K panels of 256 double-buffered in LDS
// via global_load_async_to_lds_b128 (DMA of panel k+1 overlaps WMMA on k).
// Row max/argmax packed into u64 keys, merged with ds_max_u64.
// Grid: x = m-tile (64), y = l-chunk (32), z = batch.
// ---------------------------------------------------------------------------
__global__ __launch_bounds__(256) void gemm_rowmax_kernel(
    const _Float16* __restrict__ XMT,  // (B, L, D)
    const _Float16* __restrict__ YMT,  // (B, M, D)
    float* __restrict__ part_val, int* __restrict__ part_arg,  // (B,NCHUNK,M)
    int D) {
  __shared__ _Float16 aPan[2][LSUPER][KPP];   // 2 x 128 x 528B = 132KB
  __shared__ _Float16 bPan[2][64][KPP];       // 2 x  64 x 528B =  66KB
  __shared__ unsigned long long best[64];

  const int b = blockIdx.z;
  const int m0 = blockIdx.x * 64;
  const int l0 = blockIdx.y * LSUPER;
  const int tid = threadIdx.x;
  const int lane = tid & 31;
  const int wave = tid >> 5;
  const int wl = (wave >> 1) * 32;   // 0,32,64,96
  const int wm = (wave & 1) * 32;    // 0,32

  for (int i = tid; i < 64; i += 256) best[i] = 0ULL;

  const _Float16* Xb = XMT + ((size_t)b * LPATCH + l0) * D;
  const _Float16* Yb = YMT + ((size_t)b * LPATCH + m0) * D;

  // prologue: stage panel 0
  stage_panel(&aPan[0][0][0], Xb, D, 0, tid, LSUPER);
  stage_panel(&bPan[0][0][0], Yb, D, 0, tid, 64);
  wait_async0();
  __syncthreads();

  v8f acc00 = {}, acc01 = {}, acc10 = {}, acc11 = {};
  const int NKP = D / KP;
  for (int kp = 0; kp < NKP; ++kp) {
    const int buf = kp & 1;
    if (kp + 1 < NKP) {  // overlap DMA of next panel with compute
      stage_panel(&aPan[buf ^ 1][0][0], Xb, D, kp + 1, tid, LSUPER);
      stage_panel(&bPan[buf ^ 1][0][0], Yb, D, kp + 1, tid, 64);
    }
#pragma unroll 2
    for (int kk = 0; kk < KP; kk += 32) {
      v16h a0 = lds_a_frag(&aPan[buf][0][0], wl, kk, lane);
      v16h a1 = lds_a_frag(&aPan[buf][0][0], wl + 16, kk, lane);
      v16h b0 = lds_b_frag(&bPan[buf][0][0], wm, kk, lane);
      v16h b1 = lds_b_frag(&bPan[buf][0][0], wm + 16, kk, lane);
      acc00 = wmma_f16(a0, b0, acc00);
      acc01 = wmma_f16(a0, b1, acc01);
      acc10 = wmma_f16(a1, b0, acc10);
      acc11 = wmma_f16(a1, b1, acc11);
    }
    if (kp + 1 < NKP) wait_async0();
    __syncthreads();
  }

  // epilogue: per-column max/argmax over this wave's 32 rows, then ds_max_u64
  auto update = [&](const v8f& accLo, const v8f& accHi, int colbase) {
    float bv = accLo[0];
    int r = (lane >> 4) * 8;
#pragma unroll
    for (int v = 1; v < 8; ++v)
      if (accLo[v] > bv) { bv = accLo[v]; r = (lane >> 4) * 8 + v; }
#pragma unroll
    for (int v = 0; v < 8; ++v)
      if (accHi[v] > bv) { bv = accHi[v]; r = 16 + (lane >> 4) * 8 + v; }
    float ov = __shfl_xor(bv, 16, 32);
    int orr = __shfl_xor(r, 16, 32);
    if (ov > bv || (ov == bv && orr < r)) { bv = ov; r = orr; }
    int gl = l0 + wl + r;
    unsigned int u = __float_as_uint(bv);
    u = (u & 0x80000000u) ? ~u : (u | 0x80000000u);  // sortable float key
    unsigned long long key =
        ((unsigned long long)u << 32) | (unsigned int)(~(unsigned int)gl);
    if (lane < 16) atomicMax(&best[colbase + (lane & 15)], key);
  };
  update(acc00, acc10, wm);
  update(acc01, acc11, wm + 16);
  __syncthreads();

  for (int i = tid; i < 64; i += 256) {
    unsigned long long kkey = best[i];
    unsigned int u = (unsigned int)(kkey >> 32);
    u = (u & 0x80000000u) ? (u & 0x7FFFFFFFu) : ~u;
    float val = __uint_as_float(u);
    int gl = (int)(~(unsigned int)(kkey & 0xFFFFFFFFu));
    size_t idx = ((size_t)b * NCHUNK + blockIdx.y) * LPATCH + (m0 + i);
    part_val[idx] = val;
    part_arg[idx] = gl;
  }
}

// ---------------------------------------------------------------------------
// Merge l-chunks, apply scalar normalization (R2 uses ym3's norm: reference
// bug replicated).
// ---------------------------------------------------------------------------
__global__ __launch_bounds__(256) void reduce_scale_kernel(
    const float* __restrict__ part_val, const int* __restrict__ part_arg,
    const float* __restrict__ scalX, float Nx,
    const float* __restrict__ scalY, float Ny,
    float* __restrict__ S_out, float* __restrict__ argf_out,
    int* __restrict__ arg_out) {
  int t = blockIdx.x * 256 + threadIdx.x;
  if (t >= NBATCH * LPATCH) return;
  int b = t >> 12, m = t & 4095;
  float bv = -3.402823466e38f;
  int ba = 0;
  for (int c = 0; c < NCHUNK; ++c) {
    size_t idx = ((size_t)b * NCHUNK + c) * LPATCH + m;
    float v = part_val[idx];
    if (v > bv) { bv = v; ba = part_arg[idx]; }
  }
  float mx = scalX[0] / Nx;
  float xss = fmaxf(scalX[1] - Nx * mx * mx, 1e-24f);           // sum (rs-m)^2
  float my = scalY[0] / Ny;
  float yss = fmaxf(scalY[3] - 2.f * my * scalY[2] + Ny * my * my, 1e-24f);
  float scale = 1.0f / (sqrtf(xss) * sqrtf(yss));
  S_out[t] = bv * scale;
  if (argf_out) argf_out[t] = (float)ba;
  if (arg_out) arg_out[t] = ba;
}

// ---------------------------------------------------------------------------
// T = fold(gather(unfold(ref), arg)) / 9 — direct gather per output pixel.
// ---------------------------------------------------------------------------
__global__ __launch_bounds__(256) void transfer_kernel(
    const float* __restrict__ ref, const int* __restrict__ arg,
    float* __restrict__ T, int C, int k, int pad, int stride, int Himg,
    int total) {
  int t = blockIdx.x * 256 + threadIdx.x;
  if (t >= total) return;
  int w = t % Himg;
  int h = (t / Himg) % Himg;
  int c = (t / (Himg * Himg)) % C;
  int b = t / (Himg * Himg * C);
  const float* img = ref + ((size_t)b * C + c) * Himg * Himg;
  float sum = 0.f;
  for (int p = 0; p < k; ++p) {
    int nh = h + pad - p;
    if (nh < 0 || (nh % stride)) continue;
    int i = nh / stride;
    if (i >= 64) continue;
    for (int q = 0; q < k; ++q) {
      int nw = w + pad - q;
      if (nw < 0 || (nw % stride)) continue;
      int j = nw / stride;
      if (j >= 64) continue;
      int a = arg[b * LPATCH + i * 64 + j];
      int ai = a >> 6, aj = a & 63;
      int sh = ai * stride + p - pad;
      int sw = aj * stride + q - pad;
      if ((unsigned)sh < (unsigned)Himg && (unsigned)sw < (unsigned)Himg)
        sum += img[sh * Himg + sw];
    }
  }
  T[t] = sum * (1.0f / 9.0f);
}

// ---------------------------------------------------------------------------
// Host side
// ---------------------------------------------------------------------------
extern "C" void kernel_launch(void* const* d_in, const int* in_sizes, int n_in,
                              void* d_out, int out_size, void* d_ws,
                              size_t ws_size, hipStream_t stream) {
  (void)in_sizes; (void)n_in; (void)out_size; (void)ws_size;
  const float* lrsr[3]  = {(const float*)d_in[2], (const float*)d_in[1], (const float*)d_in[0]};  // lv3,lv2,lv1
  const float* refsr[3] = {(const float*)d_in[5], (const float*)d_in[4], (const float*)d_in[3]};
  const float* reff[3]  = {(const float*)d_in[8], (const float*)d_in[7], (const float*)d_in[6]};
  float* out = (float*)d_out;

  // level params: lv3, lv2, lv1
  const int Cs[3] = {256, 128, 64};
  const int ks[3] = {3, 6, 12};
  const int ps[3] = {1, 2, 4};
  const int ss[3] = {1, 2, 4};
  const int Hs[3] = {64, 128, 256};

  // workspace carve (all offsets 256-aligned)
  char* w = (char*)d_ws;
  float* scal = (float*)(w + 0);                 // 3 levels x 4 floats
  int* arg3 = (int*)(w + 256);                   // 8192 ints
  float* rown = (float*)(w + 33024);             // up to 2*9216 floats
  float* coln = (float*)(w + 106752);            // 8192 floats
  float* pval = (float*)(w + 139520);            // 2*32*4096 floats (1MB)
  int* parg = (int*)(w + 1188096);               // 2*32*4096 ints (1MB)
  _Float16* XMT = (_Float16*)(w + 2236672);      // up to 2*4096*9216 f16
  _Float16* YMT = (_Float16*)(w + 2236672 + 150994944ULL);

  hipMemsetAsync(scal, 0, 3 * 4 * sizeof(float), stream);

  // Output offsets
  float* S_out[3]  = {out + 0, out + 8192, out + 16384};     // S3, S2, S1
  float* argf_out  = out + 24576;
  size_t t_off[3]  = {32768, 2129920, 6324224};              // T3, T2, T1
  const int t_tot[3] = {2 * 256 * 64 * 64, 2 * 128 * 128 * 128,
                        2 * 64 * 256 * 256};

  float Ntot[3];
  for (int L = 0; L < 3; ++L)
    Ntot[L] = (float)((double)NBATCH * (Cs[L] * ks[L] * ks[L]) * LPATCH);

  for (int L = 0; L < 3; ++L) {  // lv3 first: its scalars feed lv2's denom
    int C = Cs[L], k = ks[L], pad = ps[L], st = ss[L], Hi = Hs[L];
    int D = C * k * k;
    float* sc = scal + L * 4;

    rs_stats_kernel<<<NBATCH * D, 256, 0, stream>>>(refsr[L], C, k, pad, st,
                                                    Hi, rown, sc);
    lr_stats_kernel<<<NBATCH * LPATCH, 256, 0, stream>>>(lrsr[L], C, k, pad,
                                                         st, Hi, coln, sc);

    float Ninv = 1.0f / Ntot[L];
    write_panel_kernel<<<NBATCH * LPATCH, 256, 0, stream>>>(
        refsr[L], C, k, pad, st, Hi, rown, sc, Ninv, /*row norm*/ 0, XMT);
    write_panel_kernel<<<NBATCH * LPATCH, 256, 0, stream>>>(
        lrsr[L], C, k, pad, st, Hi, coln, sc, Ninv, /*col norm*/ 1, YMT);

    dim3 grid(LPATCH / 64, NCHUNK, NBATCH);
    gemm_rowmax_kernel<<<grid, 256, 0, stream>>>(XMT, YMT, pval, parg, D);

    const float* scY = (L == 1) ? (scal + 0) : sc;
    float Ny = (L == 1) ? Ntot[0] : Ntot[L];
    reduce_scale_kernel<<<(NBATCH * LPATCH + 255) / 256, 256, 0, stream>>>(
        pval, parg, sc, Ntot[L], scY, Ny, S_out[L],
        (L == 0) ? argf_out : nullptr, (L == 0) ? arg3 : nullptr);
  }

  for (int L = 0; L < 3; ++L) {
    transfer_kernel<<<(t_tot[L] + 255) / 256, 256, 0, stream>>>(
        reff[L], arg3, out + t_off[L], Cs[L], ks[L], ps[L], ss[L], Hs[L],
        t_tot[L]);
  }
}
